// RGAT_Decoder_15728170238294
// MI455X (gfx1250) — matrix operand order repaired and checked by reading, not compiled, not run on hardware
//
#include <hip/hip_runtime.h>

// ---------------- types & helpers ----------------
typedef __attribute__((ext_vector_type(16))) __bf16          v16bf;
typedef __attribute__((ext_vector_type(16))) unsigned short  v16us;
typedef __attribute__((ext_vector_type(8)))  float           v8f;

#define TP 512            // padded T (500 -> 512)
#define NEGBIG (-3.0e38f)

// native bf16 convert: lowers to gfx1250 v_cvt*bf16_f32 (vs 4-op integer rounding)
__device__ __forceinline__ unsigned short f2bf(float f) {
  return __builtin_bit_cast(unsigned short, (__bf16)f);
}
__device__ __forceinline__ float lrelu(float x) { return x > 0.f ? x : 0.2f * x; }
__device__ __forceinline__ float sigmoidf(float x) { return 1.f / (1.f + __expf(-x)); }

__device__ __forceinline__ void put8(v16us& v, int i0, uint4 q) {
  v[i0+0] = (unsigned short)(q.x & 0xffffu); v[i0+1] = (unsigned short)(q.x >> 16);
  v[i0+2] = (unsigned short)(q.y & 0xffffu); v[i0+3] = (unsigned short)(q.y >> 16);
  v[i0+4] = (unsigned short)(q.z & 0xffffu); v[i0+5] = (unsigned short)(q.z >> 16);
  v[i0+6] = (unsigned short)(q.w & 0xffffu); v[i0+7] = (unsigned short)(q.w >> 16);
}

__device__ __forceinline__ v8f wmma_bf16(v16bf a, v16bf b, v8f c) {
  return __builtin_amdgcn_wmma_f32_16x16x32_bf16(false, a, false, b, (short)0, c, false, false);
}

// A-fragment (16x32 bf16) from an f32 row in global memory.
// CDNA5 16-bit A layout: lane holds row M=lane&15; lanes<16 get K in {0..7,16..23}
// (base0=0), lanes>=16 get {8..15,24..31} (base0=8). Both 8-runs are contiguous.
__device__ __forceinline__ v16bf afrag_f32row(const float* __restrict__ row,
                                              int c32, int base0, bool ok) {
  v16us u;
  if (ok) {
    const float4* p = (const float4*)(row + c32 + base0);
    float4 f0 = p[0], f1 = p[1], f2 = p[4], f3 = p[5];
    u[0]=f2bf(f0.x);  u[1]=f2bf(f0.y);  u[2]=f2bf(f0.z);  u[3]=f2bf(f0.w);
    u[4]=f2bf(f1.x);  u[5]=f2bf(f1.y);  u[6]=f2bf(f1.z);  u[7]=f2bf(f1.w);
    u[8]=f2bf(f2.x);  u[9]=f2bf(f2.y);  u[10]=f2bf(f2.z); u[11]=f2bf(f2.w);
    u[12]=f2bf(f3.x); u[13]=f2bf(f3.y); u[14]=f2bf(f3.z); u[15]=f2bf(f3.w);
  } else {
#pragma unroll
    for (int e = 0; e < 16; ++e) u[e] = 0;
  }
  return __builtin_bit_cast(v16bf, u);
}

// A-fragment from an f32 row in LDS (row pointer = staged tile row).
__device__ __forceinline__ v16bf afrag_lds(const float* rowp, int c32, int base0) {
  v16us u;
#pragma unroll
  for (int e = 0; e < 16; ++e) {
    int kk = c32 + base0 + (e < 8 ? e : e + 8);
    u[e] = f2bf(rowp[kk]);
  }
  return __builtin_bit_cast(v16bf, u);
}

// 16x64 = (16x64)x(64x64): weights in LDS stored transposed [n][k] bf16.
__device__ __forceinline__ void gemm_k64(const v16bf a[2], const unsigned short* sW,
                                         v8f acc[4], int m16, int kb16) {
#pragma unroll
  for (int nt = 0; nt < 4; ++nt) {
    const int n = nt * 16 + m16;
#pragma unroll
    for (int c = 0; c < 2; ++c) {
      const unsigned short* p = sW + n * 64 + c * 32 + (kb16 >> 1);
      uint4 q0 = *(const uint4*)p;
      uint4 q1 = *(const uint4*)(p + 16);
      v16us u; put8(u, 0, q0); put8(u, 8, q1);
      acc[nt] = wmma_bf16(a[c], __builtin_bit_cast(v16bf, u), acc[nt]);
    }
  }
}

// 16x64 = (16x128)x(128x64)
__device__ __forceinline__ void gemm_k128(const v16bf a[4], const unsigned short* sW,
                                          v8f acc[4], int m16, int kb16) {
#pragma unroll
  for (int nt = 0; nt < 4; ++nt) {
    const int n = nt * 16 + m16;
#pragma unroll
    for (int c = 0; c < 4; ++c) {
      const unsigned short* p = sW + n * 128 + c * 32 + (kb16 >> 1);
      uint4 q0 = *(const uint4*)p;
      uint4 q1 = *(const uint4*)(p + 16);
      v16us u; put8(u, 0, q0); put8(u, 8, q1);
      acc[nt] = wmma_bf16(a[c], __builtin_bit_cast(v16bf, u), acc[nt]);
    }
  }
}

__device__ __forceinline__ void load_wT(const float* __restrict__ g, unsigned short* s,
                                        int rows, int cols, int tid, int nthr) {
  for (int i = tid; i < rows * cols; i += nthr) {
    int k = i / cols, n = i - k * cols;
    s[n * rows + k] = f2bf(g[i]);     // transposed: s[n][k]
  }
}

// ============ kernel 1: per-head projection h = inputs @ linear[h] ============
// Writes h transposed+bf16: h_t[(h*32+b)*64 + n][t] (t padded to 512, pad = 0)
// and alpha_src/alpha_tar[(h*32+b)*512 + t] = h_row . att_src/att_tar.
__global__ __launch_bounds__(256) void k1_headproj(
    const float* __restrict__ inputs, const float* __restrict__ linear,
    const float* __restrict__ att_src, const float* __restrict__ att_tar,
    unsigned short* __restrict__ h_t,
    float* __restrict__ alpha_src, float* __restrict__ alpha_tar)
{
  __shared__ unsigned short s_lin[64 * 128];   // [n][d] bf16
  __shared__ float s_as[64], s_at[64];

  const int h = blockIdx.x, b = blockIdx.y, bz = blockIdx.z;
  const int tid = threadIdx.x, w = tid >> 5, lane = tid & 31;
  const int m16 = lane & 15, hf = lane >> 4;
  const int base0 = hf * 8, kb16 = hf * 16;

  const float* lin = linear + (size_t)h * 128 * 64;
  for (int i = tid; i < 128 * 64; i += 256) {
    int d = i >> 6, n = i & 63;
    s_lin[n * 128 + d] = f2bf(lin[i]);
  }
  if (tid < 64) { s_as[tid] = att_src[h * 64 + tid]; s_at[tid] = att_tar[h * 64 + tid]; }
  __syncthreads();

  const int t0 = bz * 128 + w * 16;
  const int ta = t0 + m16;
  const bool rowok = (ta < 500);

  v16bf a[4];
  {
    const float* row = inputs + ((size_t)b * 500 + (rowok ? ta : 0)) * 128;
#pragma unroll
    for (int c = 0; c < 4; ++c) a[c] = afrag_f32row(row, c * 32, base0, rowok);
  }

  const v8f vzero = {0.f,0.f,0.f,0.f,0.f,0.f,0.f,0.f};
  v8f acc[4] = {vzero, vzero, vzero, vzero};
#pragma unroll
  for (int nt = 0; nt < 4; ++nt) {
    const int n = nt * 16 + m16;
#pragma unroll
    for (int c = 0; c < 4; ++c) {
      const unsigned short* p = s_lin + n * 128 + c * 32 + base0;
      uint4 q0 = *(const uint4*)p;
      uint4 q1 = *(const uint4*)(p + 16);
      v16us u; put8(u, 0, q0); put8(u, 8, q1);
      acc[nt] = wmma_bf16(a[c], __builtin_bit_cast(v16bf, u), acc[nt]);
    }
  }

  // store h_t transposed bf16 (D layout: elem r -> t = t0 + r + base0; 8 consecutive t)
  const size_t hb = (size_t)h * 32 + b;
#pragma unroll
  for (int nt = 0; nt < 4; ++nt) {
    const int n = nt * 16 + m16;
    uint4 q;
    q.x = (unsigned)f2bf(acc[nt][0]) | ((unsigned)f2bf(acc[nt][1]) << 16);
    q.y = (unsigned)f2bf(acc[nt][2]) | ((unsigned)f2bf(acc[nt][3]) << 16);
    q.z = (unsigned)f2bf(acc[nt][4]) | ((unsigned)f2bf(acc[nt][5]) << 16);
    q.w = (unsigned)f2bf(acc[nt][6]) | ((unsigned)f2bf(acc[nt][7]) << 16);
    *(uint4*)(h_t + (hb * 64 + n) * TP + t0 + base0) = q;   // pad rows store 0 (a=0)
  }

  // alpha reductions (sum over n of acc * att vec)
  float aps[8], apt[8];
#pragma unroll
  for (int r = 0; r < 8; ++r) { aps[r] = 0.f; apt[r] = 0.f; }
#pragma unroll
  for (int nt = 0; nt < 4; ++nt) {
    const int n = nt * 16 + m16;
    const float ws = s_as[n], wt = s_at[n];
#pragma unroll
    for (int r = 0; r < 8; ++r) { aps[r] += acc[nt][r] * ws; apt[r] += acc[nt][r] * wt; }
  }
#pragma unroll
  for (int s = 1; s < 16; s <<= 1) {
#pragma unroll
    for (int r = 0; r < 8; ++r) {
      aps[r] += __shfl_xor(aps[r], s);
      apt[r] += __shfl_xor(apt[r], s);
    }
  }
  if (m16 == 0) {   // lanes 0 (rows 0..7) and 16 (rows 8..15)
#pragma unroll
    for (int r = 0; r < 8; ++r) {
      int t = t0 + base0 + r;
      if (t < 500) {
        alpha_src[hb * TP + t] = aps[r];
        alpha_tar[hb * TP + t] = apt[r];
      }
    }
  }
}

// ============ kernel 2: per-(h,b) max over s of alpha_src ============
__global__ __launch_bounds__(256) void k2_srcmax(const float* __restrict__ alpha_src,
                                                 float* __restrict__ srcmax) {
  __shared__ float red[256];
  const int hb = blockIdx.x;
  float m = NEGBIG;
  for (int t = threadIdx.x; t < 500; t += 256) m = fmaxf(m, alpha_src[(size_t)hb * TP + t]);
  red[threadIdx.x] = m;
  __syncthreads();
  for (int s = 128; s > 0; s >>= 1) {
    if (threadIdx.x < (unsigned)s) red[threadIdx.x] = fmaxf(red[threadIdx.x], red[threadIdx.x + s]);
    __syncthreads();
  }
  if (threadIdx.x == 0) srcmax[hb] = red[0];
}

// ============ kernel 3: fused attention -> mean -> tanh -> GRU -> MLP ============
__global__ __launch_bounds__(128) void k3_att_gru_mlp(
    const float* __restrict__ inputs, const float* __restrict__ hidden,
    const float* __restrict__ bias,
    const float* __restrict__ W_hr, const float* __restrict__ W_hi, const float* __restrict__ W_hm,
    const float* __restrict__ W_ir, const float* __restrict__ b_ir,
    const float* __restrict__ W_ii, const float* __restrict__ b_ii,
    const float* __restrict__ W_in, const float* __restrict__ b_in,
    const float* __restrict__ mlp_W1, const float* __restrict__ mlp_b1,
    const float* __restrict__ mlp_W2, const float* __restrict__ mlp_b2,
    const unsigned short* __restrict__ h_t,
    const float* __restrict__ alpha_src, const float* __restrict__ alpha_tar,
    const float* __restrict__ srcmax,
    unsigned short* __restrict__ pred_bf, float* __restrict__ out_hidden)
{
  __shared__ unsigned short s_Whr[64*64], s_Whi[64*64], s_Whm[64*64];
  __shared__ unsigned short s_Wir[64*128], s_Wii[64*128], s_Win[64*128];
  __shared__ unsigned short s_W1[64*64], s_W2[64*64];
  __shared__ float s_bias[64], s_bir[64], s_bii[64], s_bin[64], s_b1[64], s_b2[64];
  __shared__ float s_stag[4][16][68];

  const int b = blockIdx.x, ty = blockIdx.y;
  const int tid = threadIdx.x, w = tid >> 5, lane = tid & 31;
  const int m16 = lane & 15, hf = lane >> 4;
  const int base0 = hf * 8, kb16 = hf * 16;

  load_wT(W_hr, s_Whr, 64, 64, tid, 128);
  load_wT(W_hi, s_Whi, 64, 64, tid, 128);
  load_wT(W_hm, s_Whm, 64, 64, tid, 128);
  load_wT(W_ir, s_Wir, 128, 64, tid, 128);
  load_wT(W_ii, s_Wii, 128, 64, tid, 128);
  load_wT(W_in, s_Win, 128, 64, tid, 128);
  load_wT(mlp_W1, s_W1, 64, 64, tid, 128);
  load_wT(mlp_W2, s_W2, 64, 64, tid, 128);
  if (tid < 64) {
    s_bias[tid] = bias[tid]; s_bir[tid] = b_ir[tid]; s_bii[tid] = b_ii[tid];
    s_bin[tid] = b_in[tid];  s_b1[tid]  = mlp_b1[tid]; s_b2[tid] = mlp_b2[tid];
  }
  __syncthreads();

  const int t0 = ty * 64 + w * 16;
  const int ta = t0 + m16;                // A-matrix row owned by this lane
  const bool rowok = (ta < 500);
  const v8f vzero = {0.f,0.f,0.f,0.f,0.f,0.f,0.f,0.f};

  // ---- flash-style single-pass attention, exact bound m_t = LR(tar_t + max_s src_s) ----
  v8f macc[4] = {vzero, vzero, vzero, vzero};       // head-mean accumulator
  for (int h = 0; h < 8; ++h) {
    const size_t hb = (size_t)h * 32 + b;
    const float srcm = srcmax[hb];
    const float tar = rowok ? alpha_tar[hb * TP + ta] : 0.f;
    const float mt = lrelu(tar + srcm);
    v8f acc[4] = {vzero, vzero, vzero, vzero};
    float psum = 0.f;
    const unsigned short* hbase = h_t + hb * 64 * TP;
    for (int s0 = 0; s0 < 500; s0 += 32) {
      const int sl = s0 + lane;
      const float srcreg = (sl < 500) ? alpha_src[hb * TP + sl] : NEGBIG;
      v16us ua;
#pragma unroll
      for (int e = 0; e < 16; ++e) {
        const int kk = base0 + (e < 8 ? e : e + 8);
        const float sv = __shfl(srcreg, kk);
        const int s = s0 + kk;
        float p = 0.f;
        if (rowok && (s != ta)) p = __expf(lrelu(tar + sv) - mt);  // s>=500 -> exp(-inf)=0
        psum += p;
        ua[e] = f2bf(p);
      }
      const v16bf pa = __builtin_bit_cast(v16bf, ua);
#pragma unroll
      for (int nt = 0; nt < 4; ++nt) {
        const unsigned short* p = hbase + (size_t)(nt * 16 + m16) * TP + s0 + base0;
        uint4 q0 = *(const uint4*)p;
        uint4 q1 = *(const uint4*)(p + 16);
        v16us ub; put8(ub, 0, q0); put8(ub, 8, q1);
        acc[nt] = wmma_bf16(pa, __builtin_bit_cast(v16bf, ub), acc[nt]);
      }
    }
    const float lall = psum + __shfl_xor(psum, 16);   // lane l: row-sum for row l&15
    float lrw[8];
#pragma unroll
    for (int r = 0; r < 8; ++r) lrw[r] = fmaxf(__shfl(lall, r + base0), 1e-30f);
#pragma unroll
    for (int nt = 0; nt < 4; ++nt)
#pragma unroll
      for (int r = 0; r < 8; ++r) macc[nt][r] += acc[nt][r] / lrw[r];
  }

  // msgs = tanh(mean_h + bias)
  v8f msgs[4];
#pragma unroll
  for (int nt = 0; nt < 4; ++nt) {
    const int n = nt * 16 + m16;
#pragma unroll
    for (int r = 0; r < 8; ++r) msgs[nt][r] = tanhf(macc[nt][r] * 0.125f + s_bias[n]);
  }

  // transpose msgs (D layout -> A layout) via LDS staging
  __syncthreads();
#pragma unroll
  for (int nt = 0; nt < 4; ++nt)
#pragma unroll
    for (int r = 0; r < 8; ++r) s_stag[w][r + base0][nt * 16 + m16] = msgs[nt][r];
  __syncthreads();
  v16bf am[2];
  am[0] = afrag_lds(&s_stag[w][m16][0], 0, base0);
  am[1] = afrag_lds(&s_stag[w][m16][0], 32, base0);

  // inputs A-fragments (16x128)
  v16bf ain[4];
  {
    const float* row = inputs + ((size_t)b * 500 + (rowok ? ta : 0)) * 128;
#pragma unroll
    for (int c = 0; c < 4; ++c) ain[c] = afrag_f32row(row, c * 32, base0, rowok);
  }

  // GRU gates
  v8f rh[4] = {vzero, vzero, vzero, vzero};  gemm_k64(am, s_Whr, rh, m16, kb16);
  v8f inr[4] = {vzero, vzero, vzero, vzero}; gemm_k128(ain, s_Wir, inr, m16, kb16);
  v8f mg[4];
#pragma unroll
  for (int nt = 0; nt < 4; ++nt) {
    const int n = nt * 16 + m16;
#pragma unroll
    for (int r = 0; r < 8; ++r) mg[nt][r] = sigmoidf(inr[nt][r] + s_bir[n] + rh[nt][r]);
  }
  v8f ihh[4] = {vzero, vzero, vzero, vzero}; gemm_k64(am, s_Whi, ihh, m16, kb16);
  v8f ini[4] = {vzero, vzero, vzero, vzero}; gemm_k128(ain, s_Wii, ini, m16, kb16);
  v8f ig[4];
#pragma unroll
  for (int nt = 0; nt < 4; ++nt) {
    const int n = nt * 16 + m16;
#pragma unroll
    for (int r = 0; r < 8; ++r) ig[nt][r] = sigmoidf(ini[nt][r] + s_bii[n] + ihh[nt][r]);
  }
  v8f mh[4] = {vzero, vzero, vzero, vzero};  gemm_k64(am, s_Whm, mh, m16, kb16);
  v8f inn[4] = {vzero, vzero, vzero, vzero}; gemm_k128(ain, s_Win, inn, m16, kb16);

  v8f hnew[4];
#pragma unroll
  for (int nt = 0; nt < 4; ++nt) {
    const int n = nt * 16 + m16;
#pragma unroll
    for (int r = 0; r < 8; ++r) {
      const int t = t0 + r + base0;                 // D-layout row -> t
      const float ng = tanhf(inn[nt][r] + s_bin[n] + mg[nt][r] * mh[nt][r]);
      const float hv = (t < 500) ? hidden[((size_t)b * 500 + t) * 64 + n] : 0.f;
      const float hn = ig[nt][r] * ng + (1.f - ig[nt][r]) * hv;
      hnew[nt][r] = hn;
      if (t < 500) out_hidden[((size_t)b * 500 + t) * 64 + n] = hn;
    }
  }

  // MLP: pred = relu(hnew@W1 + b1)@W2 + b2
  __syncthreads();
#pragma unroll
  for (int nt = 0; nt < 4; ++nt)
#pragma unroll
    for (int r = 0; r < 8; ++r) s_stag[w][r + base0][nt * 16 + m16] = hnew[nt][r];
  __syncthreads();
  v16bf ah[2];
  ah[0] = afrag_lds(&s_stag[w][m16][0], 0, base0);
  ah[1] = afrag_lds(&s_stag[w][m16][0], 32, base0);
  v8f x1[4] = {vzero, vzero, vzero, vzero};
  gemm_k64(ah, s_W1, x1, m16, kb16);
#pragma unroll
  for (int nt = 0; nt < 4; ++nt) {
    const int n = nt * 16 + m16;
#pragma unroll
    for (int r = 0; r < 8; ++r) x1[nt][r] = fmaxf(x1[nt][r] + s_b1[n], 0.f);
  }
  __syncthreads();
#pragma unroll
  for (int nt = 0; nt < 4; ++nt)
#pragma unroll
    for (int r = 0; r < 8; ++r) s_stag[w][r + base0][nt * 16 + m16] = x1[nt][r];
  __syncthreads();
  v16bf a1[2];
  a1[0] = afrag_lds(&s_stag[w][m16][0], 0, base0);
  a1[1] = afrag_lds(&s_stag[w][m16][0], 32, base0);
  v8f pr[4] = {vzero, vzero, vzero, vzero};
  gemm_k64(a1, s_W2, pr, m16, kb16);
#pragma unroll
  for (int nt = 0; nt < 4; ++nt) {
    const int n = nt * 16 + m16;
#pragma unroll
    for (int r = 0; r < 8; ++r) {
      const int t = t0 + r + base0;
      if (t < 500) pred_bf[(size_t)b * 32000 + (size_t)t * 64 + n] = f2bf(pr[nt][r] + s_b2[n]);
    }
  }
}

// ============ kernel 4: dargs = pred(32x32000) @ proj_W + proj_b; split loc|softplus ============
__global__ __launch_bounds__(256) void k4_proj(
    const unsigned short* __restrict__ pred_bf, const float* __restrict__ proj_W,
    const float* __restrict__ proj_b, float* __restrict__ d_out)
{
  __shared__ float red[2][16][17];
  const int n0 = blockIdx.x * 16;
  const int tid = threadIdx.x, w = tid >> 5, lane = tid & 31;
  const int m16 = lane & 15, hf = lane >> 4;
  const int base0 = hf * 8, kb16 = hf * 16;

  for (int i = tid; i < 2 * 16 * 17; i += 256) ((float*)red)[i] = 0.f;
  __syncthreads();

  const v8f vzero = {0.f,0.f,0.f,0.f,0.f,0.f,0.f,0.f};
  v8f acc0 = vzero, acc1 = vzero;
  const int n = n0 + m16;
  const int ncl = (n < 1000) ? n : 999;     // clamped; clamped cols discarded at store

  for (int c = w; c < 1000; c += 8) {       // 1000 K-chunks of 32, split over 8 waves
    // prefetch next chunk of the 128MB proj_W stream (global_prefetch_b8)
    if (c + 8 < 1000) {
#pragma unroll
      for (int e = 0; e < 16; e += 4) {
        const size_t sp = (size_t)((c + 8) * 32 + kb16 + e) * 1000 + ncl;
        __builtin_prefetch(&proj_W[sp], 0, 1);
      }
    }
    v16bf a0, a1;
    {
      const unsigned short* p0 = pred_bf + (size_t)m16 * 32000 + c * 32 + base0;
      uint4 q0 = *(const uint4*)p0;
      uint4 q1 = *(const uint4*)(p0 + 16);
      v16us u; put8(u, 0, q0); put8(u, 8, q1);
      a0 = __builtin_bit_cast(v16bf, u);
      const unsigned short* p1 = pred_bf + (size_t)(16 + m16) * 32000 + c * 32 + base0;
      uint4 r0 = *(const uint4*)p1;
      uint4 r1 = *(const uint4*)(p1 + 16);
      v16us v; put8(v, 0, r0); put8(v, 8, r1);
      a1 = __builtin_bit_cast(v16bf, v);
    }
    v16us ub;
#pragma unroll
    for (int e = 0; e < 16; ++e) {
      const int s = c * 32 + base0 + (e < 8 ? e : e + 8);
      ub[e] = f2bf(proj_W[(size_t)s * 1000 + ncl]);
    }
    const v16bf bb = __builtin_bit_cast(v16bf, ub);
    acc0 = wmma_bf16(a0, bb, acc0);
    acc1 = wmma_bf16(a1, bb, acc1);
  }

#pragma unroll
  for (int r = 0; r < 8; ++r) {
    atomicAdd(&red[0][r + base0][m16], acc0[r]);
    atomicAdd(&red[1][r + base0][m16], acc1[r]);
  }
  __syncthreads();

  for (int i = tid; i < 512; i += 256) {
    const int mt = i >> 8, m = (i >> 4) & 15, j = i & 15;
    const int nn = n0 + j;
    if (nn >= 1000) continue;
    const int bb = mt * 16 + m;
    const float v = red[mt][m][j] + proj_b[nn];
    if (nn < 500) {
      d_out[bb * 500 + nn] = v;                                   // loc
    } else {
      const float sp = (v > 20.f) ? v : log1pf(__expf(v));        // softplus
      d_out[16000 + bb * 500 + (nn - 500)] = sp;                  // scale
    }
  }
}

// ---------------- launcher ----------------
extern "C" void kernel_launch(void* const* d_in, const int* in_sizes, int n_in,
                              void* d_out, int out_size, void* d_ws, size_t ws_size,
                              hipStream_t stream) {
  (void)in_sizes; (void)n_in; (void)out_size; (void)ws_size;
  const float* inputs  = (const float*)d_in[0];
  const float* hidden  = (const float*)d_in[1];
  const float* linear  = (const float*)d_in[2];
  const float* bias    = (const float*)d_in[3];
  const float* att_src = (const float*)d_in[4];
  const float* att_tar = (const float*)d_in[5];
  const float* W_hr = (const float*)d_in[6];
  const float* W_hi = (const float*)d_in[7];
  const float* W_hm = (const float*)d_in[8];
  const float* W_ir = (const float*)d_in[9];
  const float* b_ir = (const float*)d_in[10];
  const float* W_ii = (const float*)d_in[11];
  const float* b_ii = (const float*)d_in[12];
  const float* W_in = (const float*)d_in[13];
  const float* b_in = (const float*)d_in[14];
  const float* mlp_W1 = (const float*)d_in[15];
  const float* mlp_b1 = (const float*)d_in[16];
  const float* mlp_W2 = (const float*)d_in[17];
  const float* mlp_b2 = (const float*)d_in[18];
  const float* proj_W = (const float*)d_in[19];
  const float* proj_b = (const float*)d_in[20];
  float* out = (float*)d_out;

  // workspace layout (bytes)
  char* ws = (char*)d_ws;
  unsigned short* h_t   = (unsigned short*)ws;                         // 8*32*64*512*2 = 16 MiB
  float* alpha_src      = (float*)(ws + 16777216);                     // 256*512*4
  float* alpha_tar      = (float*)(ws + 16777216 + 524288);
  float* srcmax         = (float*)(ws + 16777216 + 2 * 524288);        // 256*4
  unsigned short* predb = (unsigned short*)(ws + 16777216 + 2 * 524288 + 1024); // 32*32000*2

  k1_headproj<<<dim3(8, 32, 4), 256, 0, stream>>>(inputs, linear, att_src, att_tar,
                                                  h_t, alpha_src, alpha_tar);
  k2_srcmax<<<256, 256, 0, stream>>>(alpha_src, srcmax);
  k3_att_gru_mlp<<<dim3(32, 8), 128, 0, stream>>>(inputs, hidden, bias,
      W_hr, W_hi, W_hm, W_ir, b_ir, W_ii, b_ii, W_in, b_in,
      mlp_W1, mlp_b1, mlp_W2, mlp_b2,
      h_t, alpha_src, alpha_tar, srcmax, predb, out + 32000);
  k4_proj<<<63, 256, 0, stream>>>(predb, proj_W, proj_b, out);
}
